// GCNII_36215164240764
// MI455X (gfx1250) — matrix operand order repaired and checked by reading, not compiled, not run on hardware
//
#include <hip/hip_runtime.h>
#include <hip/hip_bf16.h>
#include <cmath>

// ---------------- constants (match reference) ----------------
#define NN     20000
#define MPAD   20096          // NN padded to multiple of BM (guard-free tile loads)
#define EE     320000
#define F_IN   512
#define HH     1024
#define CC     64
#define LL     8
#define ALPHA  0.1f
#define THETA  0.5f

typedef __attribute__((ext_vector_type(16))) __bf16 v16bf;
typedef __attribute__((ext_vector_type(8)))  float  v8f;

// Async global->LDS copy path (CDNA5 GLOBAL_LOAD_ASYNC_TO_LDS_*, ASYNCcnt).
// Inline asm bypasses the clang builtin's lang-address-space pointer types.
#define USE_ASYNC_LDS 1

#if USE_ASYNC_LDS
// Per-lane: copy 16 bytes from global gsrc to LDS at ldst.
// Generic pointers to __shared__ carry the LDS byte offset in addr[31:0]
// (ISA 10.2 aperture mapping), so truncation gives the VDST LDS address.
__device__ __forceinline__ void gasync_b128(const void* gsrc, void* ldst) {
    unsigned loff = (unsigned)(size_t)ldst;
    asm volatile("global_load_async_to_lds_b128 %0, %1, off"
                 :: "v"(loff), "v"(gsrc)
                 : "memory");
}
__device__ __forceinline__ void wait_async_le4() {
    asm volatile("s_wait_asynccnt 0x4" ::: "memory");
}
__device__ __forceinline__ void wait_async_le0() {
    asm volatile("s_wait_asynccnt 0x0" ::: "memory");
}
#endif

// ---------------- small utility kernels ----------------
__global__ void fill_f32(float* p, float v, int n) {
    int i = blockIdx.x * blockDim.x + threadIdx.x;
    if (i < n) p[i] = v;
}

__global__ void deg_kernel(const int* __restrict__ dst, float* deg, int e) {
    int i = blockIdx.x * blockDim.x + threadIdx.x;
    if (i < e) atomicAdd(&deg[dst[i]], 1.0f);
}

__global__ void dinv_sl_kernel(const float* __restrict__ deg, float* dinv, float* sl, int n) {
    int i = blockIdx.x * blockDim.x + threadIdx.x;
    if (i < n) {
        float r = rsqrtf(fmaxf(deg[i], 1.0f));
        dinv[i] = r;
        sl[i]   = (1.0f - ALPHA) * r * r;   // self-loop weight, pre-scaled by (1-alpha)
    }
}

__global__ void edgew_kernel(const int* __restrict__ src, const int* __restrict__ dst,
                             const float* __restrict__ dinv, float* ew, int e) {
    int i = blockIdx.x * blockDim.x + threadIdx.x;
    if (i < e) ew[i] = (1.0f - ALPHA) * dinv[src[i]] * dinv[dst[i]];
}

__global__ void cvt_bf16_kernel(const float* __restrict__ in, __bf16* __restrict__ out, int n) {
    int i = blockIdx.x * blockDim.x + threadIdx.x;
    if (i < n) out[i] = (__bf16)in[i];
}

// W [K][N] fp32 -> Wt [N][K] bf16
__global__ void transpose_cvt_kernel(const float* __restrict__ W, __bf16* __restrict__ Wt,
                                     int K, int N) {
    int i = blockIdx.x * blockDim.x + threadIdx.x;
    if (i < K * N) {
        int n = i / K, k = i - n * K;
        Wt[i] = (__bf16)W[(size_t)k * N + n];
    }
}

// z = alpha*h0 + sl[node]*h   (folds self-loop term of SpMM)
__global__ void init_z_kernel(const float* __restrict__ h0, const float* __restrict__ hin,
                              const float* __restrict__ sl, float* __restrict__ z, int n) {
    int i = blockIdx.x * blockDim.x + threadIdx.x;
    if (i < n) {
        int node = i >> 10;   // H == 1024
        z[i] = ALPHA * h0[i] + sl[node] * hin[i];
    }
}

// one block per edge: z[dst,:] += ew[e] * h[src,:]
__global__ __launch_bounds__(256)
void spmm_kernel(const int* __restrict__ src, const int* __restrict__ dst,
                 const float* __restrict__ ew, const float* __restrict__ h,
                 float* __restrict__ z) {
    int e = blockIdx.x;
    int s = src[e], d = dst[e];
    float w = ew[e];
    const float4* hs = (const float4*)(h + (size_t)s * HH);
    float* zd = z + (size_t)d * HH;
    int f = threadIdx.x;                 // 256 threads * float4 == 1024 feats
    float4 v = hs[f];
    atomicAdd(&zd[4 * f + 0], w * v.x);
    atomicAdd(&zd[4 * f + 1], w * v.y);
    atomicAdd(&zd[4 * f + 2], w * v.z);
    atomicAdd(&zd[4 * f + 3], w * v.w);
}

// ---------------- WMMA bf16 GEMM, fused epilogues ----------------
// D[M][N] = epilogue( A[Mpad][K]bf16 @ Bt[Npad][K]bf16^T )
//   MODE 0: relu(acc + bias[n])                 (h0 = relu(x@W1+b1))
//   MODE 1: relu((1-beta)*Z[m,n] + beta*acc)    (GCNII layer)
//   MODE 2: acc + bias[n]                       (logits)
// A/Bt row counts are padded to multiples of BM/BN: tile loads are guard-free,
// stores are guarded by (gm<M, gn<N) only on the tail block.
#define BM 128
#define BN 128
#define BK 32
#define LDS_STRIDE 40   // padded (elements) to reduce LDS bank conflicts

template <int MODE>
__global__ __launch_bounds__(256)
void gemm_bf16_kernel(const __bf16* __restrict__ A, const __bf16* __restrict__ Bt,
                      float* __restrict__ D, const float* __restrict__ Z,
                      const float* __restrict__ bias,
                      int M, int N, int K, float beta) {
    __shared__ __bf16 As[2][BM * LDS_STRIDE];
    __shared__ __bf16 Bs[2][BN * LDS_STRIDE];

    const int tid  = threadIdx.x;
    const int lane = tid & 31;
    const int wave = tid >> 5;       // 0..7
    const int wm   = wave >> 2;      // 0..1   (M dir, 64 rows each)
    const int wn   = wave & 3;       // 0..3   (N dir, 32 cols each)
    const int r    = lane & 15;
    const int half = lane >> 4;

    const int block_m = blockIdx.x * BM;
    const int block_n = blockIdx.y * BN;

    // each thread owns 2 of the 512 16B chunks per tile per matrix
    const int rowa = tid >> 2,          cola = (tid & 3) * 8;
    const int rowb = (tid + 256) >> 2,  colb = ((tid + 256) & 3) * 8;
    const __bf16* gA0 = A  + (size_t)(block_m + rowa) * K + cola;
    const __bf16* gA1 = A  + (size_t)(block_m + rowb) * K + colb;
    const __bf16* gB0 = Bt + (size_t)(block_n + rowa) * K + cola;
    const __bf16* gB1 = Bt + (size_t)(block_n + rowb) * K + colb;
    const int l0 = rowa * LDS_STRIDE + cola;
    const int l1 = rowb * LDS_STRIDE + colb;

    v8f acc[4][2];
#pragma unroll
    for (int a = 0; a < 4; ++a)
#pragma unroll
        for (int b = 0; b < 2; ++b)
            acc[a][b] = (v8f){0.f, 0.f, 0.f, 0.f, 0.f, 0.f, 0.f, 0.f};

    union Frag { v16bf v; uint4 q[2]; };
    const int nt = K / BK;

#if USE_ASYNC_LDS
    // ---- async double-buffered pipeline: global -> LDS via ASYNCcnt ----
    gasync_b128(gA0, &As[0][l0]);
    gasync_b128(gA1, &As[0][l1]);
    gasync_b128(gB0, &Bs[0][l0]);
    gasync_b128(gB1, &Bs[0][l1]);
    for (int t = 0; t < nt; ++t) {
        const int buf = t & 1;
        if (t + 1 < nt) {
            const int k1 = (t + 1) * BK;
            gasync_b128(gA0 + k1, &As[buf ^ 1][l0]);
            gasync_b128(gA1 + k1, &As[buf ^ 1][l1]);
            gasync_b128(gB0 + k1, &Bs[buf ^ 1][l0]);
            gasync_b128(gB1 + k1, &Bs[buf ^ 1][l1]);
            wait_async_le4();          // tile t complete, tile t+1 in flight
        } else {
            wait_async_le0();
        }
        __syncthreads();

        v16bf bfrag[2];
#pragma unroll
        for (int tn = 0; tn < 2; ++tn) {
            int n = wn * 32 + tn * 16 + r;
            Frag tf;
            tf.q[0] = *(const uint4*)(&Bs[buf][n * LDS_STRIDE + half * 8]);
            tf.q[1] = *(const uint4*)(&Bs[buf][n * LDS_STRIDE + 16 + half * 8]);
            bfrag[tn] = tf.v;
        }
#pragma unroll
        for (int tm = 0; tm < 4; ++tm) {
            int m = wm * 64 + tm * 16 + r;
            Frag tf;
            tf.q[0] = *(const uint4*)(&As[buf][m * LDS_STRIDE + half * 8]);
            tf.q[1] = *(const uint4*)(&As[buf][m * LDS_STRIDE + 16 + half * 8]);
            v16bf afrag = tf.v;
#pragma unroll
            for (int tn = 0; tn < 2; ++tn) {
                acc[tm][tn] = __builtin_amdgcn_wmma_f32_16x16x32_bf16(
                    false, afrag, false, bfrag[tn], (short)0, acc[tm][tn], false, false);
            }
        }
        if (t + 1 < nt) __syncthreads();   // protect buf^1 before next issue round
    }
#else
    // ---- fallback: register-staged double buffer ----
    uint4 rA0, rA1, rB0, rB1;
    rA0 = *(const uint4*)gA0; rA1 = *(const uint4*)gA1;
    rB0 = *(const uint4*)gB0; rB1 = *(const uint4*)gB1;
    *(uint4*)(&As[0][l0]) = rA0; *(uint4*)(&As[0][l1]) = rA1;
    *(uint4*)(&Bs[0][l0]) = rB0; *(uint4*)(&Bs[0][l1]) = rB1;
    for (int t = 0; t < nt; ++t) {
        const int buf = t & 1;
        if (t + 1 < nt) {
            const int k1 = (t + 1) * BK;
            rA0 = *(const uint4*)(gA0 + k1); rA1 = *(const uint4*)(gA1 + k1);
            rB0 = *(const uint4*)(gB0 + k1); rB1 = *(const uint4*)(gB1 + k1);
        }
        __syncthreads();

        v16bf bfrag[2];
#pragma unroll
        for (int tn = 0; tn < 2; ++tn) {
            int n = wn * 32 + tn * 16 + r;
            Frag tf;
            tf.q[0] = *(const uint4*)(&Bs[buf][n * LDS_STRIDE + half * 16]);
            tf.q[1] = *(const uint4*)(&Bs[buf][n * LDS_STRIDE + half * 16 + 8]);
            bfrag[tn] = tf.v;
        }
#pragma unroll
        for (int tm = 0; tm < 4; ++tm) {
            int m = wm * 64 + tm * 16 + r;
            Frag tf;
            tf.q[0] = *(const uint4*)(&As[buf][m * LDS_STRIDE + half * 8]);
            tf.q[1] = *(const uint4*)(&As[buf][m * LDS_STRIDE + 16 + half * 8]);
            v16bf afrag = tf.v;
#pragma unroll
            for (int tn = 0; tn < 2; ++tn) {
                acc[tm][tn] = __builtin_amdgcn_wmma_f32_16x16x32_bf16(
                    false, afrag, false, bfrag[tn], (short)0, acc[tm][tn], false, false);
            }
        }
        __syncthreads();
        if (t + 1 < nt) {
            *(uint4*)(&As[buf ^ 1][l0]) = rA0; *(uint4*)(&As[buf ^ 1][l1]) = rA1;
            *(uint4*)(&Bs[buf ^ 1][l0]) = rB0; *(uint4*)(&Bs[buf ^ 1][l1]) = rB1;
        }
    }
#endif

    // epilogue: D layout — vgpr v, half h: row = v + 8*h, col = lane&15
    // Fully unrolled so acc elements stay in VGPRs (no dynamic vector indexing).
    const bool full = (block_m + BM <= M) && (block_n + BN <= N);
    if (full) {
#pragma unroll
        for (int tm = 0; tm < 4; ++tm) {
#pragma unroll
            for (int tn = 0; tn < 2; ++tn) {
                const int gn = block_n + wn * 32 + tn * 16 + r;
                const int gm0 = block_m + wm * 64 + tm * 16 + half * 8;
                const float bv = (MODE == 1) ? 0.0f : bias[gn];
#pragma unroll
                for (int v = 0; v < 8; ++v) {
                    const size_t idx = (size_t)(gm0 + v) * N + gn;
                    float a = acc[tm][tn][v];
                    float out;
                    if (MODE == 1) {
                        out = fmaxf((1.0f - beta) * Z[idx] + beta * a, 0.0f);
                    } else if (MODE == 0) {
                        out = fmaxf(a + bv, 0.0f);
                    } else {
                        out = a + bv;
                    }
                    D[idx] = out;
                }
            }
        }
    } else {
#pragma unroll
        for (int tm = 0; tm < 4; ++tm) {
#pragma unroll
            for (int tn = 0; tn < 2; ++tn) {
                const int gn = block_n + wn * 32 + tn * 16 + r;
                const int gm0 = block_m + wm * 64 + tm * 16 + half * 8;
                const bool ok_n = (gn < N);
                const float bv = (MODE == 1 || !ok_n) ? 0.0f : bias[gn];
#pragma unroll
                for (int v = 0; v < 8; ++v) {
                    const int gm = gm0 + v;
                    if (!ok_n || gm >= M) continue;
                    const size_t idx = (size_t)gm * N + gn;
                    float a = acc[tm][tn][v];
                    float out;
                    if (MODE == 1) {
                        out = fmaxf((1.0f - beta) * Z[idx] + beta * a, 0.0f);
                    } else if (MODE == 0) {
                        out = fmaxf(a + bv, 0.0f);
                    } else {
                        out = a + bv;
                    }
                    D[idx] = out;
                }
            }
        }
    }
}

// ---------------- log-softmax over C=64, one wave per row ----------------
__global__ __launch_bounds__(256)
void logsoftmax_kernel(float* __restrict__ out, int n) {
    int row  = blockIdx.x * 8 + (threadIdx.x >> 5);
    int lane = threadIdx.x & 31;
    if (row >= n) return;
    float* p = out + (size_t)row * CC;
    float a = p[lane], b = p[lane + 32];
    float m = fmaxf(a, b);
    for (int o = 16; o > 0; o >>= 1) m = fmaxf(m, __shfl_xor(m, o, 32));
    float s = __expf(a - m) + __expf(b - m);
    for (int o = 16; o > 0; o >>= 1) s += __shfl_xor(s, o, 32);
    float lse = m + __logf(s);
    p[lane]      = a - lse;
    p[lane + 32] = b - lse;
}

// ---------------- launch ----------------
static inline size_t alignup(size_t x) { return (x + 255) & ~(size_t)255; }

extern "C" void kernel_launch(void* const* d_in, const int* in_sizes, int n_in,
                              void* d_out, int out_size, void* d_ws, size_t ws_size,
                              hipStream_t stream) {
    const float* x   = (const float*)d_in[0];
    const int*   ei  = (const int*)d_in[1];
    const float* W1  = (const float*)d_in[2];
    const float* b1  = (const float*)d_in[3];
    const float* Wg  = (const float*)d_in[4];
    const float* W2  = (const float*)d_in[5];
    const float* b2  = (const float*)d_in[6];
    float* logits    = (float*)d_out;

    const int* src = ei;
    const int* dst = ei + EE;

    // carve workspace (activation buffers padded to MPAD rows; w2t padded to BN rows)
    char* w = (char*)d_ws;
    auto carve = [&](size_t bytes) { char* p = w; w += alignup(bytes); return p; };
    float*  deg  = (float*)carve((size_t)NN * 4);
    float*  dinv = (float*)carve((size_t)NN * 4);
    float*  sl   = (float*)carve((size_t)NN * 4);
    float*  ew   = (float*)carve((size_t)EE * 4);
    float*  h0   = (float*)carve((size_t)MPAD * HH * 4);
    float*  h    = (float*)carve((size_t)MPAD * HH * 4);
    float*  z    = (float*)carve((size_t)MPAD * HH * 4);
    __bf16* zbf  = (__bf16*)carve((size_t)MPAD * HH * 2);
    __bf16* xbf  = (__bf16*)carve((size_t)MPAD * F_IN * 2);
    __bf16* w1t  = (__bf16*)carve((size_t)HH * F_IN * 2);
    __bf16* wgt  = (__bf16*)carve((size_t)LL * HH * HH * 2);
    __bf16* w2t  = (__bf16*)carve((size_t)BN * HH * 2);
    (void)ws_size; (void)n_in; (void)in_sizes; (void)out_size;

    const int T = 256;
    auto blk = [&](long n) { return (int)((n + T - 1) / T); };

    // --- graph normalization (once) ---
    fill_f32<<<blk(NN), T, 0, stream>>>(deg, 1.0f, NN);               // self-loop
    deg_kernel<<<blk(EE), T, 0, stream>>>(dst, deg, EE);
    dinv_sl_kernel<<<blk(NN), T, 0, stream>>>(deg, dinv, sl, NN);
    edgew_kernel<<<blk(EE), T, 0, stream>>>(src, dst, dinv, ew, EE);

    // --- precision conversions (once per launch) ---
    cvt_bf16_kernel<<<blk((long)NN * F_IN), T, 0, stream>>>(x, xbf, NN * F_IN);
    transpose_cvt_kernel<<<blk((long)F_IN * HH), T, 0, stream>>>(W1, w1t, F_IN, HH);
    for (int l = 0; l < LL; ++l)
        transpose_cvt_kernel<<<blk((long)HH * HH), T, 0, stream>>>(
            Wg + (size_t)l * HH * HH, wgt + (size_t)l * HH * HH, HH, HH);
    transpose_cvt_kernel<<<blk((long)HH * CC), T, 0, stream>>>(W2, w2t, HH, CC);

    const int GM = MPAD / BM;   // 157 full M-blocks, no load guards

    // --- h0 = relu(x @ W1 + b1) ---
    {
        dim3 grid(GM, HH / BN);
        gemm_bf16_kernel<0><<<grid, T, 0, stream>>>(xbf, w1t, h0, nullptr, b1,
                                                    NN, HH, F_IN, 0.0f);
    }

    // --- L GCNII layers ---
    const long nh = (long)NN * HH;
    for (int l = 0; l < LL; ++l) {
        const float beta = logf(THETA / (float)(l + 1) + 1.0f);
        const float* hin = (l == 0) ? h0 : h;
        init_z_kernel<<<blk(nh), T, 0, stream>>>(h0, hin, sl, z, (int)nh);
        spmm_kernel<<<EE, T, 0, stream>>>(src, dst, ew, hin, z);
        cvt_bf16_kernel<<<blk(nh), T, 0, stream>>>(z, zbf, (int)nh);
        dim3 grid(GM, HH / BN);
        gemm_bf16_kernel<1><<<grid, T, 0, stream>>>(zbf, wgt + (size_t)l * HH * HH, h,
                                                    z, nullptr, NN, HH, HH, beta);
    }

    // --- logits = h @ W2 + b2 ; log_softmax ---
    cvt_bf16_kernel<<<blk(nh), T, 0, stream>>>(h, zbf, (int)nh);
    {
        dim3 grid(GM, 1);
        gemm_bf16_kernel<2><<<grid, T, 0, stream>>>(zbf, w2t, logits, nullptr, b2,
                                                    NN, CC, HH, 0.0f);
    }
    logsoftmax_kernel<<<(NN + 7) / 8, T, 0, stream>>>(logits, NN);
}
